// KAN_FFT_linear_w_base_59914793779341
// MI455X (gfx1250) — matrix operand order repaired and checked by reading, not compiled, not run on hardware
//
#include <hip/hip_runtime.h>
#include <hip/hip_bf16.h>
#include <stdint.h>

// ---------------------------------------------------------------------------
// KAN FFT-linear layer as one bf16 WMMA GEMM:
//   out(4096x256) = A(4096x16640) @ W(16640x256) + bias
// K-tiles of 32: kt in [0,512): i = kt>>1, even=cos(k*x_i), odd=sin(k*x_i).
//                kt in [512,520): silu(x) features, i = (kt-512)*32 + kk.
// Main kernel: 130 iterations, each handling 4 K-tiles (two input dims i:
// cos+sin each) = 16 WMMAs, with async global->LDS double-buffered staging.
// ---------------------------------------------------------------------------

typedef __attribute__((ext_vector_type(16))) __bf16 v16bf;
typedef __attribute__((ext_vector_type(8)))  __bf16 v8bf;
typedef __attribute__((ext_vector_type(8)))  float  v8f;

#define B_DIM 4096
#define I_DIM 256
#define O_DIM 256
#define G_DIM 32
#define KT_FOURIER 512
#define KT_TOTAL   520
#define NITER      130            // 4 K-tiles per iteration
#define BM 64                     // rows per workgroup
#define BN 128                    // cols per workgroup
#define BROW 40                   // padded bf16 per B row (80B) in LDS
#define BUF_BYTES (4 * BN * BROW * 2)     // 40,960 B per buffer (4 K-tiles)
#define ITER_GBYTES (4 * O_DIM * G_DIM * 2) // 65,536 B of Wt per iteration

__device__ __forceinline__ void async_copy_b128(unsigned lds, unsigned long long ga) {
    asm volatile("global_load_async_to_lds_b128 %0, %1, off"
                 :: "v"(lds), "v"(ga) : "memory");
}

__device__ __forceinline__ void wait_async0() {
#if __has_builtin(__builtin_amdgcn_s_wait_asynccnt)
    __builtin_amdgcn_s_wait_asynccnt(0);
#else
    asm volatile("s_wait_asynccnt 0" ::: "memory");
#endif
}

// ---------------------------------------------------------------------------
// Phase 1: pack weights to bf16, layout Wt[kt][j][kk] (kk contiguous).
// ---------------------------------------------------------------------------
__global__ __launch_bounds__(256) void kan_pack_weights(
    const float* __restrict__ fc,       // (2, O, I, G)
    const float* __restrict__ mask,     // (I, O)
    const float* __restrict__ sbase,    // (I, O)
    const float* __restrict__ sfft,     // (I, O)
    __bf16* __restrict__ Wt)            // (520, 256, 32) bf16
{
    const size_t idx = (size_t)blockIdx.x * 256 + threadIdx.x;
    const size_t total = (size_t)KT_TOTAL * O_DIM * G_DIM;
    if (idx >= total) return;
    const int kk = (int)(idx & 31);
    const int j  = (int)((idx >> 5) & 255);
    const int kt = (int)(idx >> 13);
    float w;
    if (kt < KT_FOURIER) {
        const int i = kt >> 1;
        const int d = kt & 1;               // 0 = cos coeffs, 1 = sin coeffs
        const float coeff = fc[((((size_t)d * O_DIM) + j) * I_DIM + i) * G_DIM + kk];
        w = mask[i * O_DIM + j] * sfft[i * O_DIM + j] * coeff;
    } else {
        const int i = (kt - KT_FOURIER) * 32 + kk;
        w = mask[i * O_DIM + j] * sbase[i * O_DIM + j];
    }
    Wt[idx] = (__bf16)w;
}

// ---------------------------------------------------------------------------
// Phase 2: fused activation + GEMM. Block = 256 threads = 8 waves (4 M x 2 N).
// Each wave: 16 rows x 64 cols = 4 f32 accumulator tiles, 16 WMMAs/iteration.
// ---------------------------------------------------------------------------
__global__ __launch_bounds__(256) void kan_fft_gemm(
    const float*  __restrict__ x,       // (B, I)
    const __bf16* __restrict__ Wt,      // (520, 256, 32)
    const float*  __restrict__ bias,    // (O)
    float*        __restrict__ out)     // (B, O)
{
    __shared__ float  xs[BM][I_DIM + 1];        // 65,792 B
    __shared__ __bf16 Bs[2][4][BN][BROW];       // 81,920 B (dbuf x kt x col)

    const int b0  = blockIdx.x * BM;
    const int n0  = blockIdx.y * BN;
    const int tid = (int)threadIdx.x;

    // ---- precompute per-thread async staging addresses (8 x 16B chunks) ----
    // chunk q = c*256+tid in [0,2048): h = kt sub-tile, j = local col, part.
    unsigned long long ga[8];
    unsigned la[8];
    {
        const unsigned long long wb =
            (unsigned long long)(uintptr_t)Wt + (unsigned)n0 * (G_DIM * 2);
        const unsigned lb = (unsigned)(uintptr_t)(&Bs[0][0][0][0]);
#pragma unroll
        for (int c = 0; c < 8; ++c) {
            const int q = c * 256 + tid;
            const int h = q >> 9, r = q & 511, j = r >> 2, part = r & 3;
            ga[c] = wb + (unsigned)(h * (O_DIM * G_DIM * 2) + j * (G_DIM * 2) + part * 16);
            la[c] = lb + (unsigned)(h * (BN * BROW * 2) + j * (BROW * 2) + part * 16);
        }
    }

    // ---- prologue: stage iteration 0 into buf 0, then x rows into LDS ------
#pragma unroll
    for (int c = 0; c < 8; ++c) {
        async_copy_b128(la[c], ga[c]);
        ga[c] += ITER_GBYTES;
    }
    for (int it = 0; it < (BM * I_DIM / 4) / 256; ++it) {
        const int idx = tid + it * 256;
        const int r  = idx / (I_DIM / 4);
        const int c4 = idx % (I_DIM / 4);
        const float4 v = ((const float4*)(x + (size_t)(b0 + r) * I_DIM))[c4];
        xs[r][c4 * 4 + 0] = v.x;
        xs[r][c4 * 4 + 1] = v.y;
        xs[r][c4 * 4 + 2] = v.z;
        xs[r][c4 * 4 + 3] = v.w;
    }
    wait_async0();
    __syncthreads();

    const int wave = tid >> 5;
    const int lane = tid & 31;
    const int wm   = wave & 3;                  // M slice (rows wm*16..)
    const int wn   = wave >> 2;                 // N slice (cols wn*64..)
    const int row  = wm * 16 + (lane & 15);
    const int lohalf  = (lane < 16) ? 1 : 0;
    const int a_khalf = lohalf ? 0 : 8;         // A layout: lanes<16 own K{0..7,16..23}
    const int b_koff  = lohalf ? 0 : 8;         // B layout: lanes<16 own K{0..7,16..23}
    const int jloc    = wn * 64 + (lane & 15);  // local col of this lane's tile 0
    const float* xrow = &xs[row][0];

    v8f acc[4];
#pragma unroll
    for (int t = 0; t < 4; ++t) acc[t] = (v8f)0.0f;

    for (int g = 0; g < NITER; ++g) {
        const int buf = g & 1;
        // ---- issue async staging of next iteration's 4 B tiles -------------
        if (g + 1 < NITER) {                    // uniform branch
            const unsigned ldsoff = (g & 1) ? 0u : (unsigned)BUF_BYTES;
#pragma unroll
            for (int c = 0; c < 8; ++c) {
                async_copy_b128(la[c] + ldsoff, ga[c]);
                ga[c] += ITER_GBYTES;
            }
        }

        // ---- build 4 A fragments (ISA 16-bit A layout) ---------------------
        v16bf af[4];                            // cos_i0, sin_i0, cos_i1, sin_i1
        if (g < 128) {                          // uniform branch
#pragma unroll
            for (int u = 0; u < 2; ++u) {
                const float xv = xrow[2 * g + u];
                const float xr = xv * 0.15915494309189535f;     // x/(2*pi)
                const float k0 = (float)(a_khalf + 1) * xr;
#pragma unroll
                for (int p = 0; p < 16; ++p) {
                    const float dp  = (float)((p < 8) ? p : p + 8);
                    const float ang = __builtin_fmaf(dp, xr, k0);
                    af[2 * u + 0][p] = (__bf16)__builtin_amdgcn_cosf(ang);
                    af[2 * u + 1][p] = (__bf16)__builtin_amdgcn_sinf(ang);
                }
            }
        } else {
            const int ibase = (g - 128) * 128;
#pragma unroll
            for (int f = 0; f < 4; ++f) {
#pragma unroll
                for (int p = 0; p < 16; ++p) {
                    const int kk = ((p < 8) ? p : p + 8) + a_khalf;
                    const float xv = xrow[ibase + f * 32 + kk];
                    af[f][p] = (__bf16)(xv / (1.0f + __expf(-xv)));
                }
            }
        }

        // ---- 16 WMMAs: 4 K-tiles x 4 N-tiles -------------------------------
#pragma unroll
        for (int f = 0; f < 4; ++f) {
#pragma unroll
            for (int t = 0; t < 4; ++t) {
                const int jl = jloc + t * 16;
                const v8bf lo = *(const v8bf*)&Bs[buf][f][jl][b_koff];
                const v8bf hi = *(const v8bf*)&Bs[buf][f][jl][b_koff + 16];
                v16bf bfrag;
#pragma unroll
                for (int e = 0; e < 8; ++e) { bfrag[e] = lo[e]; bfrag[e + 8] = hi[e]; }
                acc[t] = __builtin_amdgcn_wmma_f32_16x16x32_bf16(
                    false, af[f], false, bfrag, (short)0, acc[t], false, false);
            }
        }

        wait_async0();          // our async stage of the other buffer done
        __syncthreads();        // all waves done reading buf / writing buf^1
    }

    // ---- epilogue: C/D layout M = v + 8*(lane>=16), N = lane%16 ------------
#pragma unroll
    for (int t = 0; t < 4; ++t) {
        const int j  = n0 + jloc + t * 16;
        const float bj = bias[j];
        const int rbase = b0 + wm * 16 + (lohalf ? 0 : 8);
#pragma unroll
        for (int v = 0; v < 8; ++v) {
            out[(size_t)(rbase + v) * O_DIM + j] = acc[t][v] + bj;
        }
    }
}

// ---------------------------------------------------------------------------
extern "C" void kernel_launch(void* const* d_in, const int* in_sizes, int n_in,
                              void* d_out, int out_size, void* d_ws, size_t ws_size,
                              hipStream_t stream) {
    (void)in_sizes; (void)n_in; (void)out_size; (void)ws_size;
    const float* x     = (const float*)d_in[0];   // (B, I)
    const float* fc    = (const float*)d_in[1];   // (2, O, I, G)
    const float* bias  = (const float*)d_in[2];   // (1, O)
    const float* mask  = (const float*)d_in[3];   // (I, O)
    const float* sbase = (const float*)d_in[4];   // (I, O)
    const float* sfft  = (const float*)d_in[5];   // (I, O)
    float*  out = (float*)d_out;
    __bf16* Wt  = (__bf16*)d_ws;                  // 520*256*32*2 = 8.5 MB

    const int pack_elems  = KT_TOTAL * O_DIM * G_DIM;
    const int pack_blocks = (pack_elems + 255) / 256;
    kan_pack_weights<<<pack_blocks, 256, 0, stream>>>(fc, mask, sbase, sfft, Wt);

    dim3 grid(B_DIM / BM, O_DIM / BN);            // 64 x 2 workgroups
    kan_fft_gemm<<<grid, 256, 0, stream>>>(x, Wt, bias, out);
}
